// MultiHeadAttention_19232863552306
// MI455X (gfx1250) — matrix operand order repaired and checked by reading, not compiled, not run on hardware
//
#include <hip/hip_runtime.h>

// ---------------------------------------------------------------------------
// MHA for MI455X (gfx1250): bf16 WMMA (16x16x32, f32 accum), flash attention.
// B=2, S=2048, D_MODEL=1024, H=16, depth=64.
// Pipeline: cvt(X)->bf16, W->W^T bf16, ping-pong double-buffered LDS-free
// WMMA GEMMs, LDS-shared flash attention with async-to-LDS staging.
// ---------------------------------------------------------------------------

typedef __attribute__((ext_vector_type(16))) __bf16 v16bf;
typedef __attribute__((ext_vector_type(8)))  float  v8f;

union Frag16 { unsigned u[8]; v16bf v; };

#define D_MODEL 1024
#define SEQ     2048
#define NH      16
#define DEPTH   64
#define BATCH   2
#define M_ROWS  (BATCH * SEQ)        // 4096

// 1/sqrt(64) * log2(e): folded into Q so softmax uses raw exp2
#define QSCALE  0.1803368801111204f

#if defined(__has_builtin)
#if __has_builtin(__builtin_amdgcn_cvt_pk_bf16_f32)
#define HAVE_CVT_PK_BF16 1
#endif
#if __has_builtin(__builtin_amdgcn_global_load_async_to_lds_b128) && \
    __has_builtin(__builtin_amdgcn_s_wait_asynccnt)
#define HAVE_ASYNC_LDS 1
#endif
#endif

#ifdef HAVE_ASYNC_LDS
// exact parameter types per compiler diagnostic:
//   arg0: v4i32 ptr in AS(1) (global), arg1: v4i32 ptr in AS(3) (LDS)
typedef int v4i_async __attribute__((vector_size(16)));
typedef __attribute__((address_space(1))) v4i_async* async_gptr;
typedef __attribute__((address_space(3))) v4i_async* async_lptr;
#endif

// pack two f32 -> packed bf16 (round-half-up + v_perm_b32: 3 VALU)
__device__ __forceinline__ unsigned pack2bf(float lo, float hi) {
#ifdef HAVE_CVT_PK_BF16
  typedef __attribute__((ext_vector_type(2))) __bf16 v2bf;
  union { v2bf h; unsigned u; } cv;
  cv.h = __builtin_amdgcn_cvt_pk_bf16_f32(lo, hi);
  return cv.u;
#else
  unsigned a = __float_as_uint(lo) + 0x8000u;
  unsigned b = __float_as_uint(hi) + 0x8000u;
  return __builtin_amdgcn_perm(b, a, 0x07060302u);   // {b.hi16, a.hi16}
#endif
}
__device__ __forceinline__ unsigned short f2bf16(float f) {
  return (unsigned short)((__float_as_uint(f) + 0x8000u) >> 16);
}

// A-fragment (16x32 bf16): per lane two contiguous 16B runs:
//   elems [h2*32 + g*8 .. +7] and [h2*32 + 16 + g*8 .. +7]
__device__ __forceinline__ Frag16 load_afrag(const unsigned short* rowp, int h2, int g) {
  Frag16 f;
  *(uint4*)&f.u[0] = *(const uint4*)(rowp + h2 * 32 + g * 8);
  *(uint4*)&f.u[4] = *(const uint4*)(rowp + h2 * 32 + 16 + g * 8);
  return f;
}
// B-fragment (32x16 bf16): per lane two contiguous 16B runs:
//   elems [h2*32 + g*8 .. +7] and [h2*32 + 16 + g*8 .. +7]
__device__ __forceinline__ Frag16 load_bfrag(const unsigned short* rowp, int h2, int g) {
  Frag16 f;
  *(uint4*)&f.u[0] = *(const uint4*)(rowp + h2 * 32 + g * 8);
  *(uint4*)&f.u[4] = *(const uint4*)(rowp + h2 * 32 + 16 + g * 8);
  return f;
}

// ---------------------------------------------------------------------------
// Prep A: fp32 -> bf16, 8 elements / thread, fully vectorized.
// ---------------------------------------------------------------------------
__global__ __launch_bounds__(256) void cvt_bf16_kernel(
    const float* __restrict__ src, unsigned short* __restrict__ dst)
{
  int i = blockIdx.x * 256 + threadIdx.x;
  const float4* s = (const float4*)src;
  float4 a = s[i * 2], b = s[i * 2 + 1];
  uint4 o;
  o.x = pack2bf(a.x, a.y); o.y = pack2bf(a.z, a.w);
  o.z = pack2bf(b.x, b.y); o.w = pack2bf(b.z, b.w);
  *(uint4*)&dst[i * 8] = o;
}

// ---------------------------------------------------------------------------
// Prep B: W [K,N] fp32 -> W^T [N,K] bf16, via 64x64 LDS tile.
// grid = (16, 16, 4); z selects which weight.
// ---------------------------------------------------------------------------
__global__ __launch_bounds__(256) void wtrans_kernel(
    const float* __restrict__ Wq, const float* __restrict__ Wk,
    const float* __restrict__ Wv, const float* __restrict__ Wo,
    unsigned short* __restrict__ Tq, unsigned short* __restrict__ Tk,
    unsigned short* __restrict__ Tv, unsigned short* __restrict__ To)
{
  const int z = blockIdx.z;
  const float* W = (z == 0) ? Wq : (z == 1) ? Wk : (z == 2) ? Wv : Wo;
  unsigned short* T = (z == 0) ? Tq : (z == 1) ? Tk : (z == 2) ? Tv : To;

  __shared__ unsigned short t[64 * 72];   // t[n][k], padded rows

  const int tid = threadIdx.x;
  const int r = tid >> 2;                 // 0..63
  const int q = (tid & 3) * 16;           // 0..48
  const int kb = blockIdx.y * 64, nb = blockIdx.x * 64;

  const float* wp = &W[(size_t)(kb + r) * D_MODEL + nb + q];
  float4 a0 = ((const float4*)wp)[0];
  float4 a1 = ((const float4*)wp)[1];
  float4 a2 = ((const float4*)wp)[2];
  float4 a3 = ((const float4*)wp)[3];
  t[(q +  0) * 72 + r] = f2bf16(a0.x); t[(q +  1) * 72 + r] = f2bf16(a0.y);
  t[(q +  2) * 72 + r] = f2bf16(a0.z); t[(q +  3) * 72 + r] = f2bf16(a0.w);
  t[(q +  4) * 72 + r] = f2bf16(a1.x); t[(q +  5) * 72 + r] = f2bf16(a1.y);
  t[(q +  6) * 72 + r] = f2bf16(a1.z); t[(q +  7) * 72 + r] = f2bf16(a1.w);
  t[(q +  8) * 72 + r] = f2bf16(a2.x); t[(q +  9) * 72 + r] = f2bf16(a2.y);
  t[(q + 10) * 72 + r] = f2bf16(a2.z); t[(q + 11) * 72 + r] = f2bf16(a2.w);
  t[(q + 12) * 72 + r] = f2bf16(a3.x); t[(q + 13) * 72 + r] = f2bf16(a3.y);
  t[(q + 14) * 72 + r] = f2bf16(a3.z); t[(q + 15) * 72 + r] = f2bf16(a3.w);
  __syncthreads();

  unsigned short* op = &T[(size_t)(nb + r) * D_MODEL + kb + q];
  *(uint4*)op       = *(const uint4*)&t[r * 72 + q];
  *(uint4*)(op + 8) = *(const uint4*)&t[r * 72 + q + 8];
}

// ---------------------------------------------------------------------------
// Kernel 1: Q/K/V projection, pure bf16, LDS-free, ping-pong k-loop.
//   grid = (N/64, M/128, 3); wave tile = 16 x 64.
// ---------------------------------------------------------------------------
__global__ __launch_bounds__(256) void qkv_gemm_kernel(
    const unsigned short* __restrict__ xq, const unsigned short* __restrict__ xk,
    const unsigned short* __restrict__ xv,
    const unsigned short* __restrict__ Tq, const unsigned short* __restrict__ Tk,
    const unsigned short* __restrict__ Tv,
    const float* __restrict__ bq, const float* __restrict__ bk,
    const float* __restrict__ bv,
    unsigned short* __restrict__ q_ws, unsigned short* __restrict__ k_ws,
    unsigned short* __restrict__ v_ws)
{
  const int which = blockIdx.z;
  const unsigned short* X  = (which == 0) ? xq : (which == 1) ? xk : xv;
  const unsigned short* WT = (which == 0) ? Tq : (which == 1) ? Tk : Tv;
  const float* bias        = (which == 0) ? bq : (which == 1) ? bk : bv;

  const int tid  = threadIdx.x;
  const int wave = tid >> 5;
  const int lane = tid & 31;
  const int g    = lane >> 4;
  const int ln   = lane & 15;

  const int n0      = blockIdx.x * 64;
  const int rowBase = blockIdx.y * 128 + wave * 16;

  const unsigned short* arow = &X[(size_t)(rowBase + ln) * D_MODEL];
  const unsigned short* brow[4];
  #pragma unroll
  for (int u = 0; u < 4; ++u)
    brow[u] = &WT[(size_t)(n0 + u * 16 + ln) * D_MODEL];

  v8f acc[4];
  #pragma unroll
  for (int u = 0; u < 4; ++u)
    #pragma unroll
    for (int r = 0; r < 8; ++r) acc[u][r] = 0.0f;

  // ping-pong double buffer: no register rotation, loads overlap WMMAs
  Frag16 afA = load_afrag(arow, 0, g);
  Frag16 bfA[4];
  #pragma unroll
  for (int u = 0; u < 4; ++u) bfA[u] = load_bfrag(brow[u], 0, g);
  Frag16 afB, bfB[4];

  for (int k0 = 0; k0 < D_MODEL - 64; k0 += 64) {
    afB = load_afrag(arow + k0 + 32, 0, g);
    #pragma unroll
    for (int u = 0; u < 4; ++u) bfB[u] = load_bfrag(brow[u] + k0 + 32, 0, g);
    #pragma unroll
    for (int u = 0; u < 4; ++u)
      acc[u] = __builtin_amdgcn_wmma_f32_16x16x32_bf16(
          false, afA.v, false, bfA[u].v, (short)0, acc[u], false, false);

    afA = load_afrag(arow + k0 + 64, 0, g);
    #pragma unroll
    for (int u = 0; u < 4; ++u) bfA[u] = load_bfrag(brow[u] + k0 + 64, 0, g);
    #pragma unroll
    for (int u = 0; u < 4; ++u)
      acc[u] = __builtin_amdgcn_wmma_f32_16x16x32_bf16(
          false, afB.v, false, bfB[u].v, (short)0, acc[u], false, false);
  }
  // epilogue: steps 960 (in A buffers) and 992
  afB = load_afrag(arow + D_MODEL - 32, 0, g);
  #pragma unroll
  for (int u = 0; u < 4; ++u) bfB[u] = load_bfrag(brow[u] + D_MODEL - 32, 0, g);
  #pragma unroll
  for (int u = 0; u < 4; ++u)
    acc[u] = __builtin_amdgcn_wmma_f32_16x16x32_bf16(
        false, afA.v, false, bfA[u].v, (short)0, acc[u], false, false);
  #pragma unroll
  for (int u = 0; u < 4; ++u)
    acc[u] = __builtin_amdgcn_wmma_f32_16x16x32_bf16(
        false, afB.v, false, bfB[u].v, (short)0, acc[u], false, false);

  // bias + scatter store in attention-friendly layouts
  #pragma unroll
  for (int u = 0; u < 4; ++u) {
    int col = n0 + u * 16 + ln;
    float bval = bias[col];
    int h = col >> 6, dd = col & 63;
    #pragma unroll
    for (int r = 0; r < 8; ++r) {
      int row = rowBase + r + g * 8;
      int b = row >> 11, s = row & (SEQ - 1);
      int bh = b * NH + h;
      float val = acc[u][r] + bval;
      if (which == 0) {
        q_ws[((size_t)bh * SEQ + s) * DEPTH + dd] = f2bf16(val * QSCALE);
      } else if (which == 1) {
        k_ws[((size_t)bh * SEQ + s) * DEPTH + dd] = f2bf16(val);
      } else {
        v_ws[((size_t)bh * DEPTH + dd) * SEQ + s] = f2bf16(val);
      }
    }
  }
}

// ---------------------------------------------------------------------------
// Kernel 2: flash attention. Block = 8 waves sharing one (b,h); each wave
// owns a 16-query tile. K/V staged per 64-key block into LDS via async
// loads, online softmax in exp2 domain.
// grid = (S/16/8, B*H), block = 256.
// ---------------------------------------------------------------------------
#define KROW 72   // padded LDS row stride (elements) for K/V/P tiles

__global__ __launch_bounds__(256) void attn_kernel(
    const unsigned short* __restrict__ q_ws,
    const unsigned short* __restrict__ k_ws,
    const unsigned short* __restrict__ v_ws,
    unsigned short* __restrict__ ctx_ws)
{
  __shared__ unsigned short ldsK[64 * KROW];          // 64 keys x 64 d
  __shared__ unsigned short ldsV[64 * KROW];          // 64 d   x 64 keys
  __shared__ unsigned short ldsP[8 * 16 * KROW];      // per-wave 16x64 P tile

  const int tid  = threadIdx.x;
  const int wave = tid >> 5;
  const int lane = tid & 31;
  const int g    = lane >> 4;
  const int ln   = lane & 15;

  const int bh = blockIdx.y;                          // 0..31
  const int q0 = (blockIdx.x * 8 + wave) * 16;        // query tile base
  unsigned short* P = &ldsP[wave * 16 * KROW];

  const unsigned short* qrow = &q_ws[((size_t)bh * SEQ + q0 + ln) * DEPTH];
  Frag16 qf[2];
  qf[0] = load_afrag(qrow, 0, g);
  qf[1] = load_afrag(qrow, 1, g);

  float m_i[8], l_i[8];
  v8f oacc[4];
  #pragma unroll
  for (int r = 0; r < 8; ++r) { m_i[r] = -1e30f; l_i[r] = 0.0f; }
  #pragma unroll
  for (int u = 0; u < 4; ++u)
    #pragma unroll
    for (int r = 0; r < 8; ++r) oacc[u][r] = 0.0f;

  for (int key0 = 0; key0 < SEQ; key0 += 64) {
    // ---- cooperative stage of K block (64x64) and V block (64x64) ----
#ifdef HAVE_ASYNC_LDS
    #pragma unroll
    for (int it = 0; it < 2; ++it) {
      int i   = tid + it * 256;
      int row = i >> 3;
      int c   = (i & 7) * 8;
      __builtin_amdgcn_global_load_async_to_lds_b128(
          (async_gptr)&k_ws[((size_t)bh * SEQ + key0 + row) * DEPTH + c],
          (async_lptr)&ldsK[row * KROW + c], 0, 0);
      __builtin_amdgcn_global_load_async_to_lds_b128(
          (async_gptr)&v_ws[((size_t)bh * DEPTH + row) * SEQ + key0 + c],
          (async_lptr)&ldsV[row * KROW + c], 0, 0);
    }
    __builtin_amdgcn_s_wait_asynccnt(0);
#else
    #pragma unroll
    for (int it = 0; it < 2; ++it) {
      int i   = tid + it * 256;
      int row = i >> 3;
      int c   = (i & 7) * 8;
      *(uint4*)&ldsK[row * KROW + c] =
          *(const uint4*)&k_ws[((size_t)bh * SEQ + key0 + row) * DEPTH + c];
      *(uint4*)&ldsV[row * KROW + c] =
          *(const uint4*)&v_ws[((size_t)bh * DEPTH + row) * SEQ + key0 + c];
    }
#endif
    if (key0 + 64 < SEQ) {   // prefetch next block (global_prefetch_b8)
      __builtin_prefetch(&k_ws[((size_t)bh * SEQ + key0 + 64 + (tid >> 2)) * DEPTH], 0, 1);
      __builtin_prefetch(&v_ws[((size_t)bh * DEPTH + (tid >> 2)) * SEQ + key0 + 64], 0, 1);
    }
    __syncthreads();

    // ---- logits (already in log2 domain): 4 tiles x 2 WMMA ----
    v8f sacc[4];
    #pragma unroll
    for (int t = 0; t < 4; ++t) {
      #pragma unroll
      for (int r = 0; r < 8; ++r) sacc[t][r] = 0.0f;
      const unsigned short* krow = &ldsK[(t * 16 + ln) * KROW];
      #pragma unroll
      for (int h2 = 0; h2 < 2; ++h2) {
        Frag16 kf = load_bfrag(krow, h2, g);
        sacc[t] = __builtin_amdgcn_wmma_f32_16x16x32_bf16(
            false, qf[h2].v, false, kf.v, (short)0, sacc[t], false, false);
      }
    }

    // ---- online softmax over the 64-key block (row = r + 8g) ----
    #pragma unroll
    for (int r = 0; r < 8; ++r) {
      float mx = fmaxf(fmaxf(sacc[0][r], sacc[1][r]),
                       fmaxf(sacc[2][r], sacc[3][r]));
      #pragma unroll
      for (int mk = 1; mk < 16; mk <<= 1)
        mx = fmaxf(mx, __shfl_xor(mx, mk, 32));
      float mnew = fmaxf(m_i[r], mx);
      float corr = __builtin_amdgcn_exp2f(m_i[r] - mnew);
      m_i[r] = mnew;
      float rs = 0.0f;
      #pragma unroll
      for (int t = 0; t < 4; ++t) {
        float p = __builtin_amdgcn_exp2f(sacc[t][r] - mnew);
        sacc[t][r] = p;
        rs += p;
      }
      #pragma unroll
      for (int mk = 1; mk < 16; mk <<= 1) rs += __shfl_xor(rs, mk, 32);
      l_i[r] = l_i[r] * corr + rs;
      #pragma unroll
      for (int u = 0; u < 4; ++u) oacc[u][r] *= corr;
    }

    // ---- transpose P (C-layout regs) -> A-layout via per-wave LDS tile ----
    #pragma unroll
    for (int t = 0; t < 4; ++t)
      #pragma unroll
      for (int r = 0; r < 8; ++r)
        P[(r + g * 8) * KROW + t * 16 + ln] = f2bf16(sacc[t][r]);
    __threadfence_block();

    Frag16 pf[2];
    pf[0] = load_afrag(&P[ln * KROW], 0, g);
    pf[1] = load_afrag(&P[ln * KROW], 1, g);

    // ---- ctx += P (16x64keys) . V (64keys x 64d): 4 tiles x 2 WMMA ----
    #pragma unroll
    for (int u = 0; u < 4; ++u) {
      const unsigned short* vrow = &ldsV[(u * 16 + ln) * KROW];
      #pragma unroll
      for (int h2 = 0; h2 < 2; ++h2) {
        Frag16 vf = load_bfrag(vrow, h2, g);
        oacc[u] = __builtin_amdgcn_wmma_f32_16x16x32_bf16(
            false, pf[h2].v, false, vf.v, (short)0, oacc[u], false, false);
      }
    }
    __syncthreads();
  }

  // ---- normalize and store ctx (bf16) in [B,S, H*64] row-major ----
  const int b = bh >> 4, h = bh & (NH - 1);
  #pragma unroll
  for (int r = 0; r < 8; ++r) {
    float inv = 1.0f / l_i[r];
    int row = b * SEQ + q0 + r + g * 8;
    #pragma unroll
    for (int u = 0; u < 4; ++u) {
      int col = h * DEPTH + u * 16 + ln;
      ctx_ws[(size_t)row * D_MODEL + col] = f2bf16(oacc[u][r] * inv);
    }
  }
}

// ---------------------------------------------------------------------------
// Kernel 3: output projection: out = ctx(bf16) @ Wo^T(bf16) + bo -> fp32.
// LDS-free, ping-pong double-buffered, same structure as qkv_gemm.
// ---------------------------------------------------------------------------
__global__ __launch_bounds__(256) void out_gemm_kernel(
    const unsigned short* __restrict__ ctx_ws,
    const unsigned short* __restrict__ To,
    const float* __restrict__ bo, float* __restrict__ out)
{
  const int tid  = threadIdx.x;
  const int wave = tid >> 5;
  const int lane = tid & 31;
  const int g    = lane >> 4;
  const int ln   = lane & 15;

  const int n0      = blockIdx.x * 64;
  const int rowBase = blockIdx.y * 128 + wave * 16;

  const unsigned short* arow = &ctx_ws[(size_t)(rowBase + ln) * D_MODEL];
  const unsigned short* brow[4];
  #pragma unroll
  for (int u = 0; u < 4; ++u)
    brow[u] = &To[(size_t)(n0 + u * 16 + ln) * D_MODEL];

  v8f acc[4];
  #pragma unroll
  for (int u = 0; u < 4; ++u)
    #pragma unroll
    for (int r = 0; r < 8; ++r) acc[u][r] = 0.0f;

  Frag16 afA = load_afrag(arow, 0, g);
  Frag16 bfA[4];
  #pragma unroll
  for (int u = 0; u < 4; ++u) bfA[u] = load_bfrag(brow[u], 0, g);
  Frag16 afB, bfB[4];

  for (int k0 = 0; k0 < D_MODEL - 64; k0 += 64) {
    afB = load_afrag(arow + k0 + 32, 0, g);
    #pragma unroll
    for (int u = 0; u < 4; ++u) bfB[u] = load_bfrag(brow[u] + k0 + 32, 0, g);
    #pragma unroll
    for (int u = 0; u < 4; ++u)
      acc[u] = __builtin_amdgcn_wmma_f32_16x16x32_bf16(
          false, afA.v, false, bfA[u].v, (short)0, acc[u], false, false);

    afA = load_afrag(arow + k0 + 64, 0, g);
    #pragma unroll
    for (int u = 0; u < 4; ++u) bfA[u] = load_bfrag(brow[u] + k0 + 64, 0, g);
    #pragma unroll
    for (int u = 0; u < 4; ++u)
      acc[u] = __builtin_amdgcn_wmma_f32_16x16x32_bf16(
          false, afB.v, false, bfB[u].v, (short)0, acc[u], false, false);
  }
  afB = load_afrag(arow + D_MODEL - 32, 0, g);
  #pragma unroll
  for (int u = 0; u < 4; ++u) bfB[u] = load_bfrag(brow[u] + D_MODEL - 32, 0, g);
  #pragma unroll
  for (int u = 0; u < 4; ++u)
    acc[u] = __builtin_amdgcn_wmma_f32_16x16x32_bf16(
        false, afA.v, false, bfA[u].v, (short)0, acc[u], false, false);
  #pragma unroll
  for (int u = 0; u < 4; ++u)
    acc[u] = __builtin_amdgcn_wmma_f32_16x16x32_bf16(
        false, afB.v, false, bfB[u].v, (short)0, acc[u], false, false);

  #pragma unroll
  for (int u = 0; u < 4; ++u) {
    int col = n0 + u * 16 + ln;
    float bval = bo[col];
    #pragma unroll
    for (int r = 0; r < 8; ++r) {
      int row = rowBase + r + g * 8;
      out[(size_t)row * D_MODEL + col] = acc[u][r] + bval;
    }
  }
}

// ---------------------------------------------------------------------------
extern "C" void kernel_launch(void* const* d_in, const int* in_sizes, int n_in,
                              void* d_out, int out_size, void* d_ws, size_t ws_size,
                              hipStream_t stream) {
  const float* q_in = (const float*)d_in[0];
  const float* k_in = (const float*)d_in[1];
  const float* v_in = (const float*)d_in[2];
  const float* Wq   = (const float*)d_in[3];
  const float* bq   = (const float*)d_in[4];
  const float* Wk   = (const float*)d_in[5];
  const float* bk   = (const float*)d_in[6];
  const float* Wv   = (const float*)d_in[7];
  const float* bv   = (const float*)d_in[8];
  const float* Wo   = (const float*)d_in[9];
  const float* bo   = (const float*)d_in[10];
  float* out = (float*)d_out;

  const size_t E = (size_t)M_ROWS * D_MODEL;            // 4M elems
  const size_t WE = (size_t)D_MODEL * D_MODEL;          // 1M elems
  unsigned short* q_ws   = (unsigned short*)d_ws;
  unsigned short* k_ws   = q_ws + E;
  unsigned short* v_ws   = k_ws + E;
  unsigned short* ctx_ws = v_ws + E;
  unsigned short* xq     = ctx_ws + E;
  unsigned short* xk     = xq + E;
  unsigned short* xv     = xk + E;
  unsigned short* Tq     = xv + E;
  unsigned short* Tk     = Tq + WE;
  unsigned short* Tv     = Tk + WE;
  unsigned short* To     = Tv + WE;

  dim3 blk(256);
  const int cvtBlocks = (int)(E / 8 / 256);             // 2048
  cvt_bf16_kernel<<<cvtBlocks, blk, 0, stream>>>(q_in, xq);
  cvt_bf16_kernel<<<cvtBlocks, blk, 0, stream>>>(k_in, xk);
  cvt_bf16_kernel<<<cvtBlocks, blk, 0, stream>>>(v_in, xv);
  wtrans_kernel<<<dim3(16, 16, 4), blk, 0, stream>>>(
      Wq, Wk, Wv, Wo, Tq, Tk, Tv, To);
  qkv_gemm_kernel<<<dim3(D_MODEL / 64, M_ROWS / 128, 3), blk, 0, stream>>>(
      xq, xk, xv, Tq, Tk, Tv, bq, bk, bv, q_ws, k_ws, v_ws);
  attn_kernel<<<dim3(SEQ / 16 / 8, BATCH * NH), blk, 0, stream>>>(
      q_ws, k_ws, v_ws, ctx_ws);
  out_gemm_kernel<<<dim3(D_MODEL / 64, M_ROWS / 128), blk, 0, stream>>>(
      ctx_ws, To, bo, out);
}
